// DeformCBAMConv_9887014715927
// MI455X (gfx1250) — matrix operand order repaired and checked by reading, not compiled, not run on hardware
//
#include <hip/hip_runtime.h>
#include <hip/hip_bf16.h>

// ---------------- problem constants ----------------
#define B_      16
#define C_IN_   128
#define C_OUT_  256
#define H_      80
#define W_      80
#define KK_     9
#define HO_     40
#define WO_     40
#define NPIX_   (HO_ * WO_)              // 1600
#define N_      (B_ * NPIX_)             // 25600 (GEMM N)
#define KDIM_   (C_IN_ * KK_)            // 1152  (GEMM K)
#define KT_     (KDIM_ / 32)             // 36 K-tiles of 32
#define NT16_   (N_ / 16)                // 1600 N-tiles of 16
#define MT16_   (C_OUT_ / 16)            // 16 M-tiles of 16
#define OUT_ELEMS_ (B_ * C_OUT_ * NPIX_) // 6,553,600

// ---------------- WMMA types ----------------
typedef __bf16 bf16_t;
typedef bf16_t v16bf __attribute__((ext_vector_type(16)));
typedef float  v8f   __attribute__((ext_vector_type(8)));

union FragU {
    uint4 q[2];
    v16bf v;
};

__device__ __forceinline__ unsigned short f32_to_bf16_rne(float f) {
    unsigned int u = __float_as_uint(f);
    u = u + 0x7FFFu + ((u >> 16) & 1u);   // round-to-nearest-even
    return (unsigned short)(u >> 16);
}

// ================= kernel 1: offset GEMM =================
// off[b][o][ho][wo] = sum_c offset_w[o][c] * x[b][c][2ho][2wo] + offset_b[o]
__global__ __launch_bounds__(256) void k_offset(const float* __restrict__ x,
                                                const float* __restrict__ ow,
                                                const float* __restrict__ ob,
                                                float* __restrict__ off) {
    int n = blockIdx.x * 256 + threadIdx.x;          // 0..25599
    int b = n / NPIX_;
    int pos = n - b * NPIX_;
    int ho = pos / WO_;
    int wo = pos - ho * WO_;
    const float* xp = x + (size_t)b * C_IN_ * H_ * W_ + (ho * 2) * W_ + (wo * 2);

    float acc[18];
#pragma unroll
    for (int o = 0; o < 18; ++o) acc[o] = ob[o];

    for (int c = 0; c < C_IN_; ++c) {
        float xv = xp[(size_t)c * H_ * W_];
#pragma unroll
        for (int o = 0; o < 18; ++o)
            acc[o] = fmaf(ow[o * C_IN_ + c], xv, acc[o]);
    }
    float* op = off + (size_t)b * 18 * NPIX_ + pos;
#pragma unroll
    for (int o = 0; o < 18; ++o) op[(size_t)o * NPIX_] = acc[o];
}

// ================= kernel 2: weights -> bf16 in WMMA-A blocked layout ===========
// Block (mt, kt) of 16(M)x32(K) stored as 512 bf16, index = lane*16 + e,
// lane = h*16 + (m%16), e<8 -> K = 8h+e ; e>=8 -> K = 16+8h+(e-8)   (ISA A layout)
__global__ __launch_bounds__(256) void k_wblk(const float* __restrict__ w,
                                              unsigned short* __restrict__ wblk) {
    int f = blockIdx.x * 256 + threadIdx.x;          // < 16*36*512 = 294912
    int blk = f >> 9;
    int r   = f & 511;
    int mt  = blk / KT_;
    int kt  = blk - mt * KT_;
    int lane = r >> 4;
    int e    = r & 15;
    int h    = lane >> 4;
    int ml   = lane & 15;
    int kl   = (e < 8) ? (8 * h + e) : (16 + 8 * h + (e - 8));
    int m = mt * 16 + ml;
    int k = kt * 32 + kl;
    wblk[f] = f32_to_bf16_rne(w[(size_t)m * KDIM_ + k]);
}

// ================= kernel 3: bilinear sample -> bf16 in WMMA-B blocked layout ====
// Block (nt, kt) of 32(K)x16(N) stored as 512 bf16, index = lane*16 + e,
// lane = h*16 + (n%16), e<8 -> K = 8h+e ; e>=8 -> K = 16+8h+(e-8)   (ISA B layout). Stores fully coalesced.
__global__ __launch_bounds__(256) void k_sample(const float* __restrict__ x,
                                                const float* __restrict__ off,
                                                unsigned short* __restrict__ sblk) {
    int f = blockIdx.x * 256 + threadIdx.x;          // < 1600*36*512 = 29,491,200
    int blk = f >> 9;
    int r   = f & 511;
    int nt  = blk / KT_;
    int kt  = blk - nt * KT_;
    int lane = r >> 4;
    int e    = r & 15;
    int h    = lane >> 4;
    int nl   = lane & 15;
    int k = kt * 32 + ((e < 8) ? (8 * h + e) : (16 + 8 * h + (e - 8)));  // 0..1151 = c*9 + kk
    int n = nt * 16 + nl;                            // 0..25599

    int c  = k / KK_;
    int kk = k - c * KK_;
    int b  = n / NPIX_;
    int pos = n - b * NPIX_;
    int ho = pos / WO_;
    int wo = pos - ho * WO_;

    const float* offb = off + ((size_t)b * 18 + kk * 2) * NPIX_;
    float dy = offb[pos];
    float dx = offb[NPIX_ + pos];

    float py = (float)(ho * 2 - 1 + (kk / 3)) + dy;
    float px = (float)(wo * 2 - 1 + (kk % 3)) + dx;

    const float* xb = x + ((size_t)b * C_IN_ + c) * (H_ * W_);
    float y0f = floorf(py), x0f = floorf(px);
    int y0 = (int)y0f, x0 = (int)x0f;
    float wy1 = py - y0f, wx1 = px - x0f;
    float wy0 = 1.f - wy1, wx0 = 1.f - wx1;

    auto tap = [&](int yi, int xi) -> float {
        if (yi < 0 || yi >= H_ || xi < 0 || xi >= W_) return 0.f;
        return xb[yi * W_ + xi];
    };
    float v = wy0 * (wx0 * tap(y0, x0)     + wx1 * tap(y0, x0 + 1)) +
              wy1 * (wx0 * tap(y0 + 1, x0) + wx1 * tap(y0 + 1, x0 + 1));

    sblk[f] = f32_to_bf16_rne(v);
}

// ================= kernel 4: main WMMA GEMM (LDS-free, blocked operands) ========
// out[m][n] = sum_k W[m][k] * S[k][n] + deform_b[m]
// Block = 256 threads = 8 waves: wm2 = wave&3 covers all of M (4 x 64),
// wn2 = wave>>2 covers 2 x 64 N columns. Wave tile 64x64 = 4x4 wmma(16x16x32).
// Fragments load directly global -> VGPR as two b128 per lane; WGP$ absorbs the
// intra-block operand reuse, L2 (192MB) holds the whole working set.
__global__ __launch_bounds__(256) void k_gemm(const unsigned short* __restrict__ wblk,
                                              const unsigned short* __restrict__ sblk,
                                              const float* __restrict__ dbias,
                                              float* __restrict__ out) {
    const int t    = threadIdx.x;
    const int wave = t >> 5;
    const int lane = t & 31;
    const int wm2  = wave & 3;        // 0..3 : m64 tile
    const int wn2  = wave >> 2;       // 0..1 : n64 tile within block
    const int half = lane >> 4;
    const int l16  = lane & 15;
    const int nb   = blockIdx.x * 2 + wn2;    // 0..399 : global n64 tile

    v8f acc[4][4];
#pragma unroll
    for (int im = 0; im < 4; ++im)
#pragma unroll
        for (int jn = 0; jn < 4; ++jn) {
            v8f z = {0.f, 0.f, 0.f, 0.f, 0.f, 0.f, 0.f, 0.f};
            acc[im][jn] = z;
        }

    for (int kt = 0; kt < KT_; ++kt) {
        FragU af[4], bfr[4];
        // A fragments: mt16 = wm2*4 + im
#pragma unroll
        for (int im = 0; im < 4; ++im) {
            const uint4* p = (const uint4*)(wblk +
                ((size_t)((wm2 * 4 + im) * KT_ + kt) << 9) + lane * 16);
            af[im].q[0] = p[0];
            af[im].q[1] = p[1];
        }
        // B fragments: nt16 = nb*4 + jn
#pragma unroll
        for (int jn = 0; jn < 4; ++jn) {
            const uint4* p = (const uint4*)(sblk +
                ((size_t)((nb * 4 + jn) * KT_ + kt) << 9) + lane * 16);
            bfr[jn].q[0] = p[0];
            bfr[jn].q[1] = p[1];
        }
        // prefetch next K-tile blocks into caches while the matrix pipe works
        if (kt + 1 < KT_) {
            __builtin_prefetch(wblk + ((size_t)((wm2 * 4) * KT_ + kt + 1) << 9) + lane * 16, 0, 1);
            __builtin_prefetch(sblk + ((size_t)((nb * 4) * KT_ + kt + 1) << 9) + lane * 16, 0, 1);
        }
        // 16 WMMAs
#pragma unroll
        for (int im = 0; im < 4; ++im)
#pragma unroll
            for (int jn = 0; jn < 4; ++jn)
                acc[im][jn] = __builtin_amdgcn_wmma_f32_16x16x32_bf16(
                    false, af[im].v, false, bfr[jn].v,
                    (short)0, acc[im][jn], false, false);
    }

    // epilogue: C/D layout VGPR r -> M = r + 8*half, N = l16
#pragma unroll
    for (int im = 0; im < 4; ++im) {
        int mb = wm2 * 64 + im * 16 + 8 * half;
#pragma unroll
        for (int jn = 0; jn < 4; ++jn) {
            int ncol = nb * 64 + jn * 16 + l16;
            int b = ncol / NPIX_;
            int pos = ncol - b * NPIX_;
            float* op = out + (size_t)b * C_OUT_ * NPIX_ + pos;
#pragma unroll
            for (int r = 0; r < 8; ++r) {
                int mm = mb + r;
                op[(size_t)mm * NPIX_] = acc[im][jn][r] + dbias[mm];
            }
        }
    }
}

// ================= kernel 5: per-channel BN stats =================
__global__ __launch_bounds__(256) void k_bnstats(const float* __restrict__ out,
                                                 float* __restrict__ stats) {
    int co = blockIdx.x;     // 0..255
    int t = threadIdx.x;
    float s1 = 0.f, s2 = 0.f;
    for (int i = t; i < N_; i += 256) {
        int b = i / NPIX_;
        int pos = i - b * NPIX_;
        float v = out[(size_t)b * C_OUT_ * NPIX_ + (size_t)co * NPIX_ + pos];
        s1 += v;
        s2 += v * v;
    }
    __shared__ float sh1[256], sh2[256];
    sh1[t] = s1; sh2[t] = s2;
    __syncthreads();
    for (int o = 128; o > 0; o >>= 1) {
        if (t < o) { sh1[t] += sh1[t + o]; sh2[t] += sh2[t + o]; }
        __syncthreads();
    }
    if (t == 0) {
        float inv = 1.f / (float)N_;
        float mean = sh1[0] * inv;
        float var = sh2[0] * inv - mean * mean;
        stats[co * 2]     = mean;
        stats[co * 2 + 1] = rsqrtf(var + 1e-5f);
    }
}

// ================= kernel 6: BN apply + SiLU (in place) =================
__global__ __launch_bounds__(256) void k_bnsilu(float* __restrict__ out,
                                                const float* __restrict__ stats,
                                                const float* __restrict__ gamma,
                                                const float* __restrict__ beta) {
    int i = blockIdx.x * 256 + threadIdx.x;
    if (i >= OUT_ELEMS_) return;
    int co = (i / NPIX_) & (C_OUT_ - 1);
    float mean = stats[co * 2];
    float rstd = stats[co * 2 + 1];
    float v = (out[i] - mean) * rstd * gamma[co] + beta[co];
    out[i] = v / (1.f + __expf(-v));
}

// ================= launcher =================
extern "C" void kernel_launch(void* const* d_in, const int* in_sizes, int n_in,
                              void* d_out, int out_size, void* d_ws, size_t ws_size,
                              hipStream_t stream) {
    const float* x        = (const float*)d_in[0];  // (16,128,80,80)
    const float* offset_w = (const float*)d_in[1];  // (18,128)
    const float* offset_b = (const float*)d_in[2];  // (18,)
    const float* deform_w = (const float*)d_in[3];  // (256,128,3,3)
    const float* deform_b = (const float*)d_in[4];  // (256,)
    const float* bn_gamma = (const float*)d_in[5];  // (256,)
    const float* bn_beta  = (const float*)d_in[6];  // (256,)
    float* out = (float*)d_out;                     // (16,256,40,40)

    // workspace carve-up
    char* ws = (char*)d_ws;
    float*          off   = (float*)ws;                                      // 1.84 MB
    unsigned short* wblk  = (unsigned short*)(ws + (size_t)2  * 1024 * 1024); // 0.59 MB
    unsigned short* sblk  = (unsigned short*)(ws + (size_t)3  * 1024 * 1024); // 59 MB
    float*          stats = (float*)(ws + (size_t)63 * 1024 * 1024);          // 2 KB

    // 1) offsets: 25600 positions
    k_offset<<<dim3(N_ / 256), dim3(256), 0, stream>>>(x, offset_w, offset_b, off);

    // 2) weights -> bf16 blocked (16*36*512 = 294912 elems)
    k_wblk<<<dim3(MT16_ * KT_ * 2), dim3(256), 0, stream>>>(deform_w, wblk);

    // 3) bilinear sample -> bf16 blocked (1600*36*512 = 29,491,200 elems)
    k_sample<<<dim3(NT16_ * KT_ * 2), dim3(256), 0, stream>>>(x, off, sblk);

    // 4) WMMA GEMM: 200 blocks x 8 waves, each wave one 64x64 output tile
    k_gemm<<<dim3(N_ / 128), dim3(256), 0, stream>>>(wblk, sblk, deform_b, out);

    // 5) BN stats per channel
    k_bnstats<<<dim3(C_OUT_), dim3(256), 0, stream>>>(out, stats);

    // 6) BN apply + SiLU
    k_bnsilu<<<dim3((OUT_ELEMS_ + 255) / 256), dim3(256), 0, stream>>>(
        out, stats, bn_gamma, bn_beta);
}
